// FMHA_39745627357185
// MI455X (gfx1250) — matrix operand order repaired and checked by reading, not compiled, not run on hardware
//
#include <hip/hip_runtime.h>

// ---------------------------------------------------------------------------
// Fused QKV projection + multi-head attention for MI455X (gfx1250, wave32).
// GEMMs run on v_wmma_f32_16x16x32_bf16; softmax/accumulation in f32.
// Kernel 1: double-buffered LDS GEMM. Kernel 2: flash attention with the
// V-tile staged by the Tensor Data Mover (TDM) including LDS row padding.
// ---------------------------------------------------------------------------

typedef __attribute__((ext_vector_type(8)))  __bf16 v8bf;
typedef __attribute__((ext_vector_type(16))) __bf16 v16bf;
typedef __attribute__((ext_vector_type(8)))  float  v8f;
typedef __attribute__((ext_vector_type(4)))  float  v4f;
typedef __attribute__((ext_vector_type(4)))  unsigned int u32x4;
typedef __attribute__((ext_vector_type(8)))  int    i32x8;
typedef __attribute__((ext_vector_type(4)))  int    i32x4;

#define HID   1024
#define NH    16
#define DH    64
#define BATCH 8
#define SEQL  1024
#define TOT   (BATCH * SEQL)
#define N3H   (3 * HID)

// Build a v16bf WMMA fragment from two 16-byte chunks (ISA 16-bit A/B layout:
// each lane holds two 8-element runs of K or N).
__device__ __forceinline__ v16bf frag2(const __bf16* p0, const __bf16* p1) {
    v8bf lo = *(const v8bf*)p0;
    v8bf hi = *(const v8bf*)p1;
    return __builtin_shufflevector(lo, hi, 0,1,2,3,4,5,6,7,8,9,10,11,12,13,14,15);
}

__device__ __forceinline__ v8f wmma_bf16(v16bf a, v16bf b, v8f c) {
    // 8 args: (neg_a, A, neg_b, B, c_mod, C, reuse_a, reuse_b)
    return __builtin_amdgcn_wmma_f32_16x16x32_bf16(false, a, false, b,
                                                   (short)0, c, false, false);
}

// ---------------------------------------------------------------------------
// Kernel 1: qkv = X @ W^T + bias     (M=8192, N=3072, K=1024)
// Double-buffered LDS tiles; output written as bf16 into ws with layout
// [sel(0/1/2)][b][h][s][d].
// ---------------------------------------------------------------------------
__global__ __launch_bounds__(256)
void qkv_gemm(const float* __restrict__ X, const float* __restrict__ W,
              const float* __restrict__ bias, __bf16* __restrict__ qkv)
{
    __shared__ __bf16 sA[2][128][40];   // [buf][m][k], +8 pad (16B rows)
    __shared__ __bf16 sB[2][32][136];   // [buf][k][n], transposed W tile

    const int m0   = blockIdx.y * 128;
    const int n0   = blockIdx.x * 128;
    const int t    = threadIdx.x;
    const int lane = t & 31;
    const int wv   = t >> 5;         // 8 waves
    const int wm   = wv & 1;         // 2 wave-rows  (64 M each)
    const int wn   = wv >> 1;        // 4 wave-cols  (32 N each)

    v8f acc[4][2];
    #pragma unroll
    for (int mt = 0; mt < 4; ++mt)
        #pragma unroll
        for (int nt = 0; nt < 2; ++nt)
            acc[mt][nt] = (v8f)0.0f;

    const int lk  = (t & 7) * 4;     // k offset within 32-wide tile
    const int lr  = t >> 3;          // row 0..31 (stride 32 over 128 rows)
    const int hi8 = (lane & 16) ? 8 : 0;

    v4f ax[4], wx[4];

    // Prologue: stage k-tile 0 into buffer 0.
    #pragma unroll
    for (int i = 0; i < 4; ++i) {
        ax[i] = *(const v4f*)(X + (size_t)(m0 + lr + 32 * i) * HID + lk);
        wx[i] = *(const v4f*)(W + (size_t)(n0 + lr + 32 * i) * HID + lk);
    }
    #pragma unroll
    for (int i = 0; i < 4; ++i) {
        const int r = lr + 32 * i;
        sA[0][r][lk + 0] = (__bf16)ax[i].x;
        sA[0][r][lk + 1] = (__bf16)ax[i].y;
        sA[0][r][lk + 2] = (__bf16)ax[i].z;
        sA[0][r][lk + 3] = (__bf16)ax[i].w;
        sB[0][lk + 0][r] = (__bf16)wx[i].x;
        sB[0][lk + 1][r] = (__bf16)wx[i].y;
        sB[0][lk + 2][r] = (__bf16)wx[i].z;
        sB[0][lk + 3][r] = (__bf16)wx[i].w;
    }

    for (int it = 0; it < HID / 32; ++it) {
        __syncthreads();
        const int cur = it & 1;
        const bool more = (it + 1) < (HID / 32);

        // Issue next tile's global loads early so they overlap the WMMAs.
        if (more) {
            const int kk = (it + 1) * 32;
            #pragma unroll
            for (int i = 0; i < 4; ++i) {
                ax[i] = *(const v4f*)(X + (size_t)(m0 + lr + 32 * i) * HID + kk + lk);
                wx[i] = *(const v4f*)(W + (size_t)(n0 + lr + 32 * i) * HID + kk + lk);
            }
        }

        #pragma unroll
        for (int mt = 0; mt < 4; ++mt) {
            const int m = wm * 64 + mt * 16 + (lane & 15);
            // A layout: lane = M row; K chunks {0-7,16-23} / {8-15,24-31}
            v16bf a = frag2(&sA[cur][m][hi8], &sA[cur][m][hi8 + 16]);
            #pragma unroll
            for (int nt = 0; nt < 2; ++nt) {
                const int nb = wn * 32 + nt * 16;
                // B layout: lane = N col; K chunks {0-7,16-23} / {8-15,24-31}
                v16bf b;
                #pragma unroll
                for (int s = 0; s < 16; ++s)
                    b[s] = sB[cur][(s & 7) + hi8 + (s & 8) * 2][nb + (lane & 15)];
                acc[mt][nt] = wmma_bf16(a, b, acc[mt][nt]);
            }
        }

        // Convert + store the next tile into the other buffer (no barrier
        // needed: distinct buffer; next iteration's barrier orders readers).
        if (more) {
            const int nxt = cur ^ 1;
            #pragma unroll
            for (int i = 0; i < 4; ++i) {
                const int r = lr + 32 * i;
                sA[nxt][r][lk + 0] = (__bf16)ax[i].x;
                sA[nxt][r][lk + 1] = (__bf16)ax[i].y;
                sA[nxt][r][lk + 2] = (__bf16)ax[i].z;
                sA[nxt][r][lk + 3] = (__bf16)ax[i].w;
                sB[nxt][lk + 0][r] = (__bf16)wx[i].x;
                sB[nxt][lk + 1][r] = (__bf16)wx[i].y;
                sB[nxt][lk + 2][r] = (__bf16)wx[i].z;
                sB[nxt][lk + 3][r] = (__bf16)wx[i].w;
            }
        }
    }

    // Epilogue: bias in f32, convert to bf16, scatter to [sel][b][h][s][d]
    const int hi = lane >> 4;
    const int nn = lane & 15;
    #pragma unroll
    for (int mt = 0; mt < 4; ++mt) {
        #pragma unroll
        for (int nt = 0; nt < 2; ++nt) {
            const int gn  = n0 + wn * 32 + nt * 16 + nn;   // 0..3071
            const int sel = gn >> 10;
            const int h   = (gn & 1023) >> 6;
            const int d   = gn & 63;
            const float bi = bias[gn];
            #pragma unroll
            for (int r = 0; r < 8; ++r) {
                const int gm = m0 + wm * 64 + mt * 16 + r + 8 * hi; // token
                const float v = acc[mt][nt][r] + bi;
                const size_t off =
                    ((((size_t)sel * BATCH + (gm >> 10)) * NH + h) * SEQL +
                     (gm & 1023)) * DH + d;
                qkv[off] = (__bf16)v;
            }
        }
    }
}

// ---------------------------------------------------------------------------
// Kernel 2: flash attention. One block = (b, h, 64-query slab).
// 4 waves x 16 query rows; 64-key steps; online softmax in f32.
// V tile is staged by the Tensor Data Mover with built-in LDS row padding
// (32 DWORDs data + 4 DWORDs pad == the sV[64][72] layout); K is transposed
// manually by the VALU in parallel with the DMA.
// ---------------------------------------------------------------------------
__global__ __launch_bounds__(128)
void attn(const __bf16* __restrict__ qkv, float* __restrict__ out)
{
    __shared__ __bf16 sKT[64][72];       // [d][key]  (K transposed)
    __shared__ __bf16 sV [64][72];       // [key][d]  (filled by TDM)
    __shared__ __bf16 sP [4][16][72];    // per-wave P tile (accum->A reshape)

    const int bh = blockIdx.x;           // 0..127
    const int b  = bh >> 4;
    const int h  = bh & 15;
    const int q0 = blockIdx.y * 64;
    const int t  = threadIdx.x;
    const int lane = t & 31;
    const int wv   = t >> 5;

    const __bf16* Qb = qkv + (((size_t)0 * BATCH + b) * NH + h) * SEQL * DH;
    const __bf16* Kb = qkv + (((size_t)1 * BATCH + b) * NH + h) * SEQL * DH;
    const __bf16* Vb = qkv + (((size_t)2 * BATCH + b) * NH + h) * SEQL * DH;

    // LDS byte offset of sV (generic LDS addresses carry the offset in the
    // low 32 bits).
    const unsigned ldsV = (unsigned)(uintptr_t)&sV[0][0];

    // Q fragments: held in registers for the whole kernel (layout matches A).
    const int c0   = (lane & 16) ? 8 : 0;
    const int qrow = q0 + wv * 16 + (lane & 15);
    v16bf qf[2];
    #pragma unroll
    for (int ds = 0; ds < 2; ++ds) {
        const __bf16* p = Qb + (size_t)qrow * DH + ds * 32 + c0;
        qf[ds] = frag2(p, p + 16);
    }

    float rowmax[8], rowsum[8];
    v8f acc_o[4];
    #pragma unroll
    for (int r = 0; r < 8; ++r) { rowmax[r] = -1e30f; rowsum[r] = 0.0f; }
    #pragma unroll
    for (int dt = 0; dt < 4; ++dt) acc_o[dt] = (v8f)0.0f;

    const int lrow = t >> 1;             // loader: key row 0..63
    const int ldh  = (t & 1) * 32;       // loader: d-half

    for (int kv = 0; kv < SEQL; kv += 64) {
        __syncthreads();                 // protect sKT/sV reuse

        // --- TDM: DMA the 64x64 bf16 V tile into sV with row padding. ---
        if (wv == 0) {
            const unsigned long long ga =
                (unsigned long long)(uintptr_t)(Vb + (size_t)kv * DH);
            u32x4 g0;
            g0[0] = 1u;                               // count=1 (valid D#)
            g0[1] = ldsV;                             // lds_addr
            g0[2] = (unsigned)ga;                     // global_addr[31:0]
            g0[3] = (unsigned)(ga >> 32) | 0x80000000u; // addr[56:32]|type=2
            i32x8 g1;
            g1[0] = 0x07110000;   // data_size=2B, pad_en, every 32dw pad 4dw
            g1[1] = 64 << 16;     // tensor_dim0 = 64 elements
            g1[2] = 1024 << 16;   // tensor_dim1 = 1024 rows
            g1[3] = 64 << 16;     // tile_dim0 = 64
            g1[4] = 64;           // tile_dim1 = 64
            g1[5] = 64;           // tensor_dim0_stride = 64
            g1[6] = 0;
            g1[7] = 0;
            i32x4 gz4 = (i32x4)0; // groups 2/3 unused (2D tensor)
            i32x8 gz8 = (i32x8)0; // 6-arg toolchain form: extra group, zeroed
            __builtin_amdgcn_tensor_load_to_lds(g0, g1, gz4, gz4, gz8, 0);
        }

        // --- Meanwhile: manual K transpose into sKT. ---
        #pragma unroll
        for (int j = 0; j < 4; ++j) {
            v8bf kvec = *(const v8bf*)(Kb + (size_t)(kv + lrow) * DH + ldh + j * 8);
            #pragma unroll
            for (int e = 0; e < 8; ++e)
                sKT[ldh + j * 8 + e][lrow] = kvec[e];
        }

        if (wv == 0) __builtin_amdgcn_s_wait_tensorcnt(0);
        __syncthreads();

        // S = Q * K^T  (16 queries x 64 keys per wave)
        v8f s[4];
        #pragma unroll
        for (int nt = 0; nt < 4; ++nt) s[nt] = (v8f)0.0f;
        #pragma unroll
        for (int ds = 0; ds < 2; ++ds) {
            #pragma unroll
            for (int nt = 0; nt < 4; ++nt) {
                // B layout: lane = key (N col); d chunks {0-7,16-23} / {8-15,24-31}
                const __bf16* kp = Kb + (size_t)(kv + nt * 16 + (lane & 15)) * DH + ds * 32 + c0;
                v16bf bfr = frag2(kp, kp + 16);
                s[nt] = wmma_bf16(qf[ds], bfr, s[nt]);
            }
        }

        // Online softmax. Row m = r + 8*(lane>=16); its 16 columns per tile
        // live in the 16 lanes of this half -> reduce with shfl_xor 1/2/4/8.
        const float scale = 0.125f;      // 1/sqrt(64)
        #pragma unroll
        for (int r = 0; r < 8; ++r) {
            float mx = -1e30f;
            #pragma unroll
            for (int nt = 0; nt < 4; ++nt) {
                s[nt][r] *= scale;
                mx = fmaxf(mx, s[nt][r]);
            }
            #pragma unroll
            for (int msk = 1; msk <= 8; msk <<= 1)
                mx = fmaxf(mx, __shfl_xor(mx, msk, 32));
            const float mnew = fmaxf(rowmax[r], mx);
            const float corr = __expf(rowmax[r] - mnew);
            rowmax[r] = mnew;
            float ps = 0.0f;
            #pragma unroll
            for (int nt = 0; nt < 4; ++nt) {
                const float p = __expf(s[nt][r] - mnew);
                s[nt][r] = p;
                ps += p;
            }
            #pragma unroll
            for (int msk = 1; msk <= 8; msk <<= 1)
                ps += __shfl_xor(ps, msk, 32);
            rowsum[r] = rowsum[r] * corr + ps;
            #pragma unroll
            for (int dt = 0; dt < 4; ++dt)
                acc_o[dt][r] *= corr;    // rescale running output
        }

        // Reshape P (C/D layout) -> A layout via LDS, as bf16.
        const int hi = lane >> 4;
        const int nn = lane & 15;
        #pragma unroll
        for (int nt = 0; nt < 4; ++nt)
            #pragma unroll
            for (int r = 0; r < 8; ++r)
                sP[wv][r + 8 * hi][nt * 16 + nn] = (__bf16)s[nt][r];
        __syncthreads();

        // O += P * V
        #pragma unroll
        for (int ks = 0; ks < 2; ++ks) {
            const __bf16* prow = &sP[wv][lane & 15][ks * 32 + c0];
            v16bf pa = frag2(prow, prow + 16);
            #pragma unroll
            for (int dt = 0; dt < 4; ++dt) {
                v16bf bv;
                #pragma unroll
                for (int s = 0; s < 16; ++s)
                    bv[s] = sV[ks * 32 + (s & 7) + c0 + (s & 8) * 2][dt * 16 + (lane & 15)];
                acc_o[dt] = wmma_bf16(pa, bv, acc_o[dt]);
            }
        }
    }

    // Final normalize + store f32 output (T, HID)
    const int hi = lane >> 4;
    const int nn = lane & 15;
    #pragma unroll
    for (int dt = 0; dt < 4; ++dt)
        #pragma unroll
        for (int r = 0; r < 8; ++r) {
            const int q = q0 + wv * 16 + r + 8 * hi;
            const float o = acc_o[dt][r] / rowsum[r];
            out[(size_t)(b * SEQL + q) * HID + h * DH + dt * 16 + nn] = o;
        }
}

// ---------------------------------------------------------------------------
extern "C" void kernel_launch(void* const* d_in, const int* in_sizes, int n_in,
                              void* d_out, int out_size, void* d_ws, size_t ws_size,
                              hipStream_t stream) {
    (void)in_sizes; (void)n_in; (void)out_size; (void)ws_size;
    const float* X    = (const float*)d_in[0];   // hidden_states (8192,1024)
    const float* W    = (const float*)d_in[1];   // Wqkv (3072,1024)
    const float* bias = (const float*)d_in[2];   // Bqkv (3072,)
    // d_in[3] = cu_seqlens, d_in[4] = max_s : fixed uniform sizes, unused.

    __bf16* qkv = (__bf16*)d_ws;                 // 3*8*16*1024*64 bf16 = 50 MB
    float*  out = (float*)d_out;

    dim3 g1(N3H / 128, TOT / 128);               // 24 x 64 blocks
    qkv_gemm<<<g1, 256, 0, stream>>>(X, W, bias, qkv);

    dim3 g2(BATCH * NH, SEQL / 64);              // 128 x 16 blocks
    attn<<<g2, 128, 0, stream>>>(qkv, out);
}
